// MultiHeadAttention_21801253994841
// MI455X (gfx1250) — compile-verified, matrix operand order checked by reading, not run on hardware
//
#include <hip/hip_runtime.h>
#include <hip/hip_bf16.h>

// ---------------------------------------------------------------------------
// Multi-head causal attention with RoPE for MI455X (gfx1250, wave32, WMMA).
// Pipeline: f32->bf16 convert -> fused QKV GEMM (+RoPE, V stored transposed)
//           -> per-tile flash attention -> output projection GEMM (f32 out).
// Matmuls: v_wmma_f32_16x16x32_bf16. GEMM staging: GLOBAL_LOAD_ASYNC_TO_LDS
// (CDNA5 async memory->LDS path, ASYNCcnt), double-buffered LDS.
// ---------------------------------------------------------------------------

typedef __bf16 bf16_t;
typedef __attribute__((ext_vector_type(16))) __bf16 v16bf;
typedef __attribute__((ext_vector_type(8)))  __bf16 v8bf;
typedef __attribute__((ext_vector_type(8)))  float  v8f;

#define NHEAD 16
#define DHEAD 64
#define SEQT  2048
#define BATCH 2
#define CDIM  1024
#define MTOT  (BATCH * SEQT)   // 4096

// Build a 16x32 bf16 A/B fragment for wmma_f32_16x16x32_bf16.
// Per-lane layout (ISA 7.12.2): row/col = lane%16; kbase = (lane>=16)?8:0;
// elements 0..7 hold K=kbase..kbase+7, elements 8..15 hold K=kbase+16..kbase+23.
// Both chunks are contiguous in a row-major [.,K] matrix -> two 16B loads.
__device__ __forceinline__ v16bf make_frag(const bf16_t* p_lo, const bf16_t* p_hi) {
  v8bf lo = *(const v8bf*)p_lo;
  v8bf hi = *(const v8bf*)p_hi;
  return __builtin_shufflevector(lo, hi, 0,1,2,3,4,5,6,7,8,9,10,11,12,13,14,15);
}

__device__ __forceinline__ v8f wmma_bf16(v16bf a, v16bf b, v8f c) {
  return __builtin_amdgcn_wmma_f32_16x16x32_bf16(false, a, false, b, (short)0, c, false, false);
}

// Async memory->LDS copy, 16 bytes per lane (VGLOBAL GLOBAL_LOAD_ASYNC_TO_LDS_B128,
// tracked by ASYNCcnt). lds_addr: wave-relative LDS byte offset (low 32 bits of a
// generic pointer to __shared__); gaddr: 64-bit global address.
__device__ __forceinline__ void async_ld_lds_b128(const void* lds_ptr, const void* gptr) {
  unsigned lds_a = (unsigned)(uintptr_t)lds_ptr;
  unsigned long long ga = (unsigned long long)(uintptr_t)gptr;
  asm volatile("global_load_async_to_lds_b128 %0, %1, off" :: "v"(lds_a), "v"(ga) : "memory");
}
__device__ __forceinline__ void wait_asynccnt0() {
  asm volatile("s_wait_asynccnt 0x0" ::: "memory");
}

// ---------------------------------------------------------------------------
// f32 -> bf16 conversion
// ---------------------------------------------------------------------------
__global__ void cvt_bf16_kernel(const float* __restrict__ x, bf16_t* __restrict__ y, int n) {
  int i = blockIdx.x * blockDim.x + threadIdx.x;
  if (i < n) y[i] = (bf16_t)x[i];
}

// ---------------------------------------------------------------------------
// RoPE tables: cos/sin of t * 10000^(-2d/64), d in [0,32). Halves duplicate.
// ---------------------------------------------------------------------------
__global__ void rope_tab_kernel(float* __restrict__ ct, float* __restrict__ st) {
  int idx = blockIdx.x * blockDim.x + threadIdx.x;   // T*32
  if (idx >= SEQT * 32) return;
  int t = idx >> 5, d = idx & 31;
  float inv = powf(10000.f, -((float)(2 * d)) / 64.f);
  float ang = (float)t * inv;
  ct[idx] = cosf(ang);
  st[idx] = sinf(ang);
}

// ---------------------------------------------------------------------------
// Tiled WMMA GEMM:  y[m,n] = sum_k A[m,k] * W[n,k]   (i.e. y = A @ W^T)
// Block: 128 threads (4 waves) -> 128x64 output tile; each wave 32x64.
// Double-buffered LDS, async staging, one barrier per K-step (32).
// mode 0: Q proj (+RoPE) -> bf16 [B,H,T,D]
// mode 1: K proj (+RoPE) -> bf16 [B,H,T,D]
// mode 2: V proj          -> bf16 [B,H,D,T]  (transposed for PV B-fragments)
// mode 3: out  proj       -> f32  [M,C] (d_out)
// ---------------------------------------------------------------------------
__global__ __launch_bounds__(128)
void gemm_wmma_kernel(const bf16_t* __restrict__ Ab,
                      const bf16_t* __restrict__ W0,
                      const bf16_t* __restrict__ W1,
                      const bf16_t* __restrict__ W2,
                      bf16_t* __restrict__ Qo, bf16_t* __restrict__ Ko,
                      bf16_t* __restrict__ Vt, float* __restrict__ Fo,
                      const float* __restrict__ cosT, const float* __restrict__ sinT,
                      int mode_base)
{
  // K-stride 40 elements (80B): 16B-aligned chunks, staggered banks.
  __shared__ bf16_t As[2][128 * 40];   // 2 x 10240 B
  __shared__ bf16_t Bs[2][64 * 40];    // 2 x  5120 B

  const int mode = mode_base + blockIdx.z;
  const bf16_t* W = (mode == 1) ? W1 : (mode == 2) ? W2 : W0;

  const int m0   = blockIdx.x * 128;
  const int n0   = blockIdx.y * 64;
  const int tid  = threadIdx.x;
  const int wid  = tid >> 5;
  const int lane = tid & 31;
  const int half = lane >> 4;
  const int c    = lane & 15;
  const int kb   = half * 8;

  // Async staging: each lane DMAs 16B chunks straight into LDS (no VGPR bounce).
  auto stage = [&](int kt, int s) {
    #pragma unroll
    for (int it = 0; it < 4; ++it) {                 // A: 128 rows x 32 cols
      int idx = tid + it * 128;
      int row = idx >> 2, col = (idx & 3) * 8;
      async_ld_lds_b128(&As[s][row * 40 + col],
                        &Ab[(size_t)(m0 + row) * CDIM + kt + col]);
    }
    #pragma unroll
    for (int it = 0; it < 2; ++it) {                 // B: 64 rows x 32 cols
      int idx = tid + it * 128;
      int row = idx >> 2, col = (idx & 3) * 8;
      async_ld_lds_b128(&Bs[s][row * 40 + col],
                        &W[(size_t)(n0 + row) * CDIM + kt + col]);
    }
  };

  v8f acc[2][4] = {};
  stage(0, 0);

  int s = 0;
  for (int kt = 0; kt < CDIM; kt += 32, s ^= 1) {
    wait_asynccnt0();                      // our async stage of buf s landed in LDS
    __syncthreads();                       // everyone's did; buf s^1 free for reuse
    if (kt + 32 < CDIM) stage(kt + 32, s ^ 1);   // DMA next tile during compute

    const int ar = wid * 32 + c;
    // preload all fragments, then issue the WMMA chain (loads overlap wmma)
    v16bf a0 = make_frag(&As[s][ar * 40 + kb],        &As[s][ar * 40 + kb + 16]);
    v16bf a1 = make_frag(&As[s][(ar + 16) * 40 + kb], &As[s][(ar + 16) * 40 + kb + 16]);
    v16bf bfr[4];
    #pragma unroll
    for (int j = 0; j < 4; ++j)
      bfr[j] = make_frag(&Bs[s][(j * 16 + c) * 40 + kb], &Bs[s][(j * 16 + c) * 40 + kb + 16]);
    #pragma unroll
    for (int j = 0; j < 4; ++j) {
      acc[0][j] = wmma_bf16(a0, bfr[j], acc[0][j]);
      acc[1][j] = wmma_bf16(a1, bfr[j], acc[1][j]);
    }
  }

  // epilogue
  const int h = n0 >> 6;   // head (N tile == one head for modes 0..2)
  #pragma unroll
  for (int r = 0; r < 2; ++r) {
    #pragma unroll
    for (int i = 0; i < 8; ++i) {
      const int m = m0 + wid * 32 + r * 16 + half * 8 + i;  // C/D layout: row = i + 8*half
      const int b = m >> 11;                                // m / 2048
      const int t = m & (SEQT - 1);
      if (mode <= 1) {
        bf16_t* O = (mode == 0) ? Qo : Ko;
        const size_t base = (((size_t)(b * NHEAD + h)) * SEQT + t) * DHEAD;
        #pragma unroll
        for (int j = 0; j < 2; ++j) {             // d0 < 32; partner at d0+32 in acc[r][j+2]
          const int d0 = j * 16 + c;
          const float cv = cosT[t * 32 + d0];
          const float sv = sinT[t * 32 + d0];
          const float v0 = acc[r][j][i], v2 = acc[r][j + 2][i];
          O[base + d0]      = (bf16_t)(v0 * cv - v2 * sv);
          O[base + d0 + 32] = (bf16_t)(v2 * cv + v0 * sv);
        }
      } else if (mode == 2) {
        const size_t hb = ((size_t)(b * NHEAD + h)) * DHEAD;
        #pragma unroll
        for (int j = 0; j < 4; ++j) {
          const int d = j * 16 + c;
          Vt[(hb + d) * SEQT + t] = (bf16_t)acc[r][j][i];
        }
      } else {
        #pragma unroll
        for (int j = 0; j < 4; ++j)
          Fo[(size_t)m * CDIM + n0 + j * 16 + c] = acc[r][j][i];
      }
    }
  }
}

// ---------------------------------------------------------------------------
// Flash attention: one wave per (16-row tile, head, batch).
// S = Q K^T (2 WMMAs per 16-col tile, K-dim=64), online softmax with 16-lane
// shuffles, P bounced through LDS to A-fragment layout, O += P V (4 WMMAs).
// V fragments are loaded before the softmax VALU work to hide global latency.
// ---------------------------------------------------------------------------
__global__ __launch_bounds__(32)
void flash_attn_kernel(const bf16_t* __restrict__ Q, const bf16_t* __restrict__ K,
                       const bf16_t* __restrict__ Vt, bf16_t* __restrict__ Ao)
{
  __shared__ bf16_t Ps[16 * 40];

  const int rt = blockIdx.x, h = blockIdx.y, b = blockIdx.z;
  const int lane = threadIdx.x & 31;
  const int half = lane >> 4, c = lane & 15, kb = half * 8;
  const int row0 = rt * 16;

  const size_t headQK = ((size_t)(b * NHEAD + h)) * SEQT;   // rows of Q/K [*,64]
  const size_t headV  = ((size_t)(b * NHEAD + h)) * DHEAD;  // rows of Vt  [*,T]

  // Q fragments (rows row0+c, d halves 0..31 / 32..63), RoPE already applied
  const bf16_t* qrow = &Q[(headQK + row0 + c) * DHEAD];
  const v16bf qf0 = make_frag(qrow + kb,      qrow + kb + 16);
  const v16bf qf1 = make_frag(qrow + 32 + kb, qrow + 32 + kb + 16);

  float mrow[8], lrow[8];
  v8f oacc[4] = {};
  #pragma unroll
  for (int i = 0; i < 8; ++i) { mrow[i] = -3.0e38f; lrow[i] = 0.f; }

  for (int st = 0; st <= rt; st += 2) {    // two 16-col score tiles -> K=32 for PV
    // ---- scores ----
    v8f s0 = {}, s1 = {};
    {
      const bf16_t* kr0 = &K[(headQK + (size_t)(st * 16 + c)) * DHEAD];
      s0 = wmma_bf16(qf0, make_frag(kr0 + kb,      kr0 + kb + 16),      s0);
      s0 = wmma_bf16(qf1, make_frag(kr0 + 32 + kb, kr0 + 32 + kb + 16), s0);
      const bf16_t* kr1 = &K[(headQK + (size_t)((st + 1) * 16 + c)) * DHEAD];
      s1 = wmma_bf16(qf0, make_frag(kr1 + kb,      kr1 + kb + 16),      s1);
      s1 = wmma_bf16(qf1, make_frag(kr1 + 32 + kb, kr1 + 32 + kb + 16), s1);
    }

    // ---- V fragments (issue early: overlap with softmax VALU) ----
    v16bf vf[4];
    #pragma unroll
    for (int j = 0; j < 4; ++j) {
      const bf16_t* vr = &Vt[(headV + (size_t)(j * 16 + c)) * SEQT + st * 16];
      vf[j] = make_frag(vr + kb, vr + kb + 16);
    }

    // ---- online softmax ----
    #pragma unroll
    for (int i = 0; i < 8; ++i) {
      const int trow = row0 + half * 8 + i;
      const int c0 = st * 16 + c;
      const float v0 = (c0      <= trow) ? s0[i] * 0.125f : -3.0e38f;  // scale = 1/sqrt(64)
      const float v1 = (c0 + 16 <= trow) ? s1[i] * 0.125f : -3.0e38f;
      float mx = fmaxf(v0, v1);
      #pragma unroll
      for (int msk = 8; msk >= 1; msk >>= 1) mx = fmaxf(mx, __shfl_xor(mx, msk, 16));
      const float mnew  = fmaxf(mrow[i], mx);
      const float alpha = __expf(mrow[i] - mnew);
      const float e0 = __expf(v0 - mnew);
      const float e1 = __expf(v1 - mnew);
      float sum = e0 + e1;
      #pragma unroll
      for (int msk = 8; msk >= 1; msk >>= 1) sum += __shfl_xor(sum, msk, 16);
      lrow[i] = lrow[i] * alpha + sum;
      mrow[i] = mnew;
      #pragma unroll
      for (int j = 0; j < 4; ++j) oacc[j][i] *= alpha;
      Ps[(half * 8 + i) * 40 + c]      = (bf16_t)e0;   // P in [16 x 32] row-major
      Ps[(half * 8 + i) * 40 + c + 16] = (bf16_t)e1;
    }
    __syncthreads();   // single wave: compiler fence + ds waits

    // ---- O += P V ----
    const v16bf pf = make_frag(&Ps[c * 40 + kb], &Ps[c * 40 + kb + 16]);
    #pragma unroll
    for (int j = 0; j < 4; ++j)
      oacc[j] = wmma_bf16(pf, vf[j], oacc[j]);
    __syncthreads();   // protect Ps against next iteration's stores
  }

  // normalize and store to [B,T,C] bf16
  const size_t obase = (size_t)b * SEQT * CDIM;
  #pragma unroll
  for (int i = 0; i < 8; ++i) {
    const int t = row0 + half * 8 + i;
    const float inv = 1.f / lrow[i];
    #pragma unroll
    for (int j = 0; j < 4; ++j)
      Ao[obase + (size_t)t * CDIM + h * DHEAD + j * 16 + c] = (bf16_t)(oacc[j][i] * inv);
  }
}

// ---------------------------------------------------------------------------
// Host-side launch
// ---------------------------------------------------------------------------
extern "C" void kernel_launch(void* const* d_in, const int* in_sizes, int n_in,
                              void* d_out, int out_size, void* d_ws, size_t ws_size,
                              hipStream_t stream)
{
  (void)in_sizes; (void)n_in; (void)out_size; (void)ws_size;

  const float* x  = (const float*)d_in[0];
  const float* Wq = (const float*)d_in[1];
  const float* Wk = (const float*)d_in[2];
  const float* Wv = (const float*)d_in[3];
  const float* Wo = (const float*)d_in[4];
  float* out = (float*)d_out;

  char* ws = (char*)d_ws;
  size_t off = 0;
  auto alloc = [&](size_t bytes) -> char* {
    char* p = ws + off;
    off += (bytes + 255) & ~(size_t)255;
    return p;
  };

  bf16_t* xb    = (bf16_t*)alloc((size_t)MTOT * CDIM * 2);
  bf16_t* wqb   = (bf16_t*)alloc((size_t)CDIM * CDIM * 2);
  bf16_t* wkb   = (bf16_t*)alloc((size_t)CDIM * CDIM * 2);
  bf16_t* wvb   = (bf16_t*)alloc((size_t)CDIM * CDIM * 2);
  bf16_t* wob   = (bf16_t*)alloc((size_t)CDIM * CDIM * 2);
  bf16_t* Qb    = (bf16_t*)alloc((size_t)MTOT * CDIM * 2);   // [B,H,T,D]
  bf16_t* Kb    = (bf16_t*)alloc((size_t)MTOT * CDIM * 2);   // [B,H,T,D]
  bf16_t* Vtb   = (bf16_t*)alloc((size_t)MTOT * CDIM * 2);   // [B,H,D,T]
  bf16_t* attnb = (bf16_t*)alloc((size_t)MTOT * CDIM * 2);   // [B,T,C]
  float*  cosT  = (float*)alloc((size_t)SEQT * 32 * 4);
  float*  sinT  = (float*)alloc((size_t)SEQT * 32 * 4);

  // 1) f32 -> bf16
  {
    int n = MTOT * CDIM;
    cvt_bf16_kernel<<<(n + 255) / 256, 256, 0, stream>>>(x, xb, n);
    int nw = CDIM * CDIM;
    cvt_bf16_kernel<<<(nw + 255) / 256, 256, 0, stream>>>(Wq, wqb, nw);
    cvt_bf16_kernel<<<(nw + 255) / 256, 256, 0, stream>>>(Wk, wkb, nw);
    cvt_bf16_kernel<<<(nw + 255) / 256, 256, 0, stream>>>(Wv, wvb, nw);
    cvt_bf16_kernel<<<(nw + 255) / 256, 256, 0, stream>>>(Wo, wob, nw);
  }

  // 2) RoPE tables
  rope_tab_kernel<<<(SEQT * 32 + 255) / 256, 256, 0, stream>>>(cosT, sinT);

  // 3) fused QKV projections (+RoPE on Q,K; V transposed)
  gemm_wmma_kernel<<<dim3(MTOT / 128, CDIM / 64, 3), 128, 0, stream>>>(
      xb, wqb, wkb, wvb, Qb, Kb, Vtb, nullptr, cosT, sinT, /*mode_base=*/0);

  // 4) causal flash attention
  flash_attn_kernel<<<dim3(SEQT / 16, NHEAD, BATCH), 32, 0, stream>>>(Qb, Kb, Vtb, attnb);

  // 5) output projection -> f32 d_out
  gemm_wmma_kernel<<<dim3(MTOT / 128, CDIM / 64, 1), 128, 0, stream>>>(
      attnb, wob, nullptr, nullptr, nullptr, nullptr, nullptr, out, cosT, sinT, /*mode_base=*/3);
}
